// CSPN_61881888800839
// MI455X (gfx1250) — matrix operand order repaired and checked by reading, not compiled, not run on hardware
//
#include <hip/hip_runtime.h>
#include <hip/hip_bf16.h>
#include <stdint.h>

// ---------------------------------------------------------------------------
// CSPN on gfx1250: implicit-GEMM 3x3 convs via v_wmma_f32_16x16x32_bf16.
// Activations live in spatially padded NHWC bf16 buffers (zero borders), so
// the GEMM K-loop is fully unrolled, branch-free, and every load uses a
// compile-time immediate offset. Fixed size: B=4, C=64, H=256, W=512.
// ---------------------------------------------------------------------------

#define DEVFN __device__ __forceinline__

constexpr int Bn = 4, Hh = 256, Ww = 512;
constexpr int Wp = Ww + 2, Hp = Hh + 2;                 // padded geometry
constexpr size_t Ptot   = (size_t)Bn * Hh * Ww;         // 524288 interior px
constexpr size_t PSlice = (size_t)Hp * Wp;              // padded batch slice
constexpr size_t PadTot = (size_t)Bn * PSlice;
constexpr float BN_EPS = 1e-5f;
constexpr float GK_EPS = 1e-6f;

typedef __attribute__((ext_vector_type(16))) __bf16 v16bf;
typedef __attribute__((ext_vector_type(8)))  float  v8f;

union Frag32B { uint4 q[2]; v16bf v; };

DEVFN unsigned short f2bf(float f) {
  unsigned u = __float_as_uint(f);
  unsigned r = u + 0x7FFFu + ((u >> 16) & 1u);   // round-to-nearest-even
  return (unsigned short)(r >> 16);
}
DEVFN float bf2f(unsigned short h) { return __uint_as_float(((unsigned)h) << 16); }

// ---------------------------------------------------------------------------
// 0) zero the 1-px borders of the padded activation buffers (NH 64ch,
//    CT 96ch, Z1 64ch). Interior is fully overwritten by later kernels.
// ---------------------------------------------------------------------------
__global__ void k_zero_border(unsigned short* __restrict__ nh,
                              unsigned short* __restrict__ ct,
                              unsigned short* __restrict__ z1) {
  size_t idx = (size_t)blockIdx.x * 256 + threadIdx.x;
  if (idx >= PadTot) return;
  size_t rem = idx % PSlice;
  int y = (int)(rem / Wp), x = (int)(rem % Wp);
  if (y != 0 && y != Hp - 1 && x != 0 && x != Wp - 1) return;
#pragma unroll
  for (int c = 0; c < 64; ++c) nh[idx * 64 + c] = 0;
#pragma unroll
  for (int c = 0; c < 96; ++c) ct[idx * 96 + c] = 0;
#pragma unroll
  for (int c = 0; c < 64; ++c) z1[idx * 64 + c] = 0;
}

// ---------------------------------------------------------------------------
// 1) fout NCHW f32 -> padded NHWC bf16 (64ch) + ct input (96ch, ch64..95 = 0)
//    nhI/ctI are pre-offset to interior pixel (0,0) of batch 0.
// ---------------------------------------------------------------------------
__global__ void k_convert(const float* __restrict__ fout,
                          unsigned short* __restrict__ nhI,
                          unsigned short* __restrict__ ctI) {
  size_t idx = (size_t)blockIdx.x * 256 + threadIdx.x;
  if (idx >= Ptot * 64) return;
  size_t p = idx >> 6;
  int c = (int)(idx & 63);
  size_t plane = (size_t)Hh * Ww;
  int b = (int)(p / plane);
  size_t rem = p - (size_t)b * plane;
  int y = (int)(rem / Ww), x = (int)(rem % Ww);
  size_t pp = ((size_t)b * Hp + y) * Wp + x;            // padded pixel index
  float v = fout[((size_t)b * 64 + c) * plane + rem];
  unsigned short h = f2bf(v);
  nhI[pp * 64 + c] = h;
  ctI[pp * 96 + c] = h;
  if (c < 32) ctI[pp * 96 + 64 + c] = 0;                // zero pad ch 64..95
}

// ---------------------------------------------------------------------------
// 2) weight swizzle: OIHW f32 -> B-fragment-ordered bf16
//    frag[(t*NT+nt)*32 + lane][j]; lane holds n = nt*16 + lane%16,
//    k = (lane<16 ? 0 : 16) + j; chunk t -> tap = t/CPT, c0 = (t%CPT)*32
// ---------------------------------------------------------------------------
__global__ void k_swizzle(const float* __restrict__ w,
                          unsigned short* __restrict__ dst,
                          int Cout, int Cin, int CPT, int NT) {
  int t  = blockIdx.x / NT;
  int nt = blockIdx.x % NT;
  int lane = threadIdx.x & 31;
  int j    = threadIdx.x >> 5;          // 0..15
  int k   = ((lane < 16) ? 0 : 16) + j;
  int tap = t / CPT;
  int c0  = (t - tap * CPT) * 32;
  int cin = c0 + k;
  int n   = nt * 16 + (lane & 15);
  float v = 0.f;
  if (n < Cout && cin < Cin) v = w[((size_t)n * Cin + cin) * 9 + tap];
  dst[((size_t)(t * NT + nt) * 32 + lane) * 16 + j] = f2bf(v);
}

// ---------------------------------------------------------------------------
// 3) implicit-GEMM conv3x3 (SAME via zero borders): srcI padded NHWC bf16,
//    dstI NHWC bf16 (padded iff DP). 4 waves/block; each wave computes a
//    16-pixel x NT*16-channel tile. K-loop fully unrolled: 9 taps x CPT
//    chunks of 32; all load offsets are compile-time immediates.
// ---------------------------------------------------------------------------
template <int NT, int CPT, bool DP>
__global__ __launch_bounds__(128) void k_gemm(
    const unsigned short* __restrict__ srcI,
    const unsigned short* __restrict__ wfrag,
    unsigned short* __restrict__ dstI, int Cdst,
    float* __restrict__ stats) {
  constexpr int Cpad = CPT * 32;
  __shared__ float ssum[64];
  __shared__ float ssq[64];
  const int tid  = threadIdx.x;
  const int lane = tid & 31;
  const int wave = tid >> 5;
  if (stats) {
    if (tid < 64) { ssum[tid] = 0.f; ssq[tid] = 0.f; }
    __syncthreads();
  }
  const int XT = Ww / 64;                       // 8 x-tiles per row
  int xt = blockIdx.x % XT;
  int y  = (blockIdx.x / XT) % Hh;
  int b  = blockIdx.x / (XT * Hh);
  const int m   = lane & 15;
  const int x   = xt * 64 + wave * 16 + m;      // this lane's A-row pixel
  const int hib = (lane < 16) ? 0 : 16;

  const unsigned short* ap0 =
      srcI + (((size_t)b * Hp + y) * Wp + x) * Cpad + hib;
  const unsigned short* wb = wfrag + (size_t)lane * 16;

  v8f acc[NT];
#pragma unroll
  for (int i = 0; i < NT; ++i)
#pragma unroll
    for (int e = 0; e < 8; ++e) acc[i][e] = 0.f;

#pragma unroll
  for (int tap = 0; tap < 9; ++tap) {
    const int dy = tap / 3 - 1, dx = tap % 3 - 1;   // compile-time constants
#pragma unroll
    for (int cc = 0; cc < CPT; ++cc) {
      const int t = tap * CPT + cc;
      Frag32B au;
      const unsigned short* ap = ap0 + (dy * Wp + dx) * Cpad + cc * 32;
      au.q[0] = *(const uint4*)(ap);        // K {c0+hib .. +7}
      au.q[1] = *(const uint4*)(ap + 8);    // K {c0+hib+8 .. +15}
#pragma unroll
      for (int nt = 0; nt < NT; ++nt) {
        Frag32B bu;
        const unsigned short* wq = wb + (size_t)(t * NT + nt) * 512;
        bu.q[0] = *(const uint4*)(wq);
        bu.q[1] = *(const uint4*)(wq + 8);
        acc[nt] = __builtin_amdgcn_wmma_f32_16x16x32_bf16(
            false, au.v, false, bu.v, (short)0, acc[nt], false, false);
      }
    }
  }

  // Epilogue: store D (bf16) + per-channel partial BN stats
  constexpr int    DRow = DP ? Wp : Ww;
  constexpr size_t DSl  = DP ? (size_t)Hp * Wp : (size_t)Hh * Ww;
#pragma unroll
  for (int nt = 0; nt < NT; ++nt) {
    int n = nt * 16 + m;
    float s = 0.f, s2 = 0.f;
#pragma unroll
    for (int v = 0; v < 8; ++v) {
      int M = v + ((lane < 16) ? 0 : 8);
      float d = acc[nt][v];
      s += d; s2 += d * d;
      int px = xt * 64 + wave * 16 + M;
      dstI[((size_t)b * DSl + (size_t)y * DRow + px) * Cdst + n] = f2bf(d);
    }
    if (stats) { atomicAdd(&ssum[n], s); atomicAdd(&ssq[n], s2); }
  }
  if (stats) {
    __syncthreads();
    if (tid < NT * 16) {
      atomicAdd(&stats[tid], ssum[tid]);
      atomicAdd(&stats[64 + tid], ssq[tid]);
    }
  }
}

// ---------------------------------------------------------------------------
// 4) BN finalize: stats[0..63]=sum, [64..127]=sumsq -> [128..]=a, [192..]=b
// ---------------------------------------------------------------------------
__global__ void k_bn_finalize(float* __restrict__ stats,
                              const float* __restrict__ g,
                              const float* __restrict__ be,
                              int C, float cnt) {
  int c = threadIdx.x;
  if (c >= C) return;
  float m = stats[c] / cnt;
  float v = stats[64 + c] / cnt - m * m;
  float a = g[c] * rsqrtf(v + BN_EPS);
  stats[128 + c] = a;
  stats[192 + c] = be[c] - m * a;
}

// 5) apply BN+ReLU in place on the interior of a padded 64ch NHWC buffer
__global__ void k_bn_relu(unsigned short* __restrict__ zI,
                          const float* __restrict__ stats) {
  size_t idx = (size_t)blockIdx.x * 256 + threadIdx.x;
  if (idx >= Ptot * 64) return;
  size_t p = idx >> 6;
  int c = (int)(idx & 63);
  size_t plane = (size_t)Hh * Ww;
  int b = (int)(p / plane);
  size_t rem = p - (size_t)b * plane;
  int y = (int)(rem / Ww), x = (int)(rem % Ww);
  size_t a = (((size_t)b * Hp + y) * Wp + x) * 64 + c;
  float v = stats[128 + c] * bf2f(zI[a]) + stats[192 + c];
  zI[a] = f2bf(v > 0.f ? v : 0.f);
}

// ---------------------------------------------------------------------------
// 6) GenKernel normalize: BN(z2) -> L1 normalize -> insert wmid at center
//    z2 unpadded NHWC; output wk[p][NK+1] f32
// ---------------------------------------------------------------------------
template <int NK>
__global__ void k_genkernel(const unsigned short* __restrict__ z2, int Cdst,
                            const float* __restrict__ stats,
                            float* __restrict__ wk) {
  size_t p = (size_t)blockIdx.x * 256 + threadIdx.x;
  if (p >= Ptot) return;
  float w[NK];
  float s = 0.f;
#pragma unroll
  for (int i = 0; i < NK; ++i) {
    float v = stats[128 + i] * bf2f(z2[p * Cdst + i]) + stats[192 + i];
    w[i] = v;
    s += fabsf(v);
  }
  float inv = 1.f / (s + GK_EPS);
  float sm = 0.f;
#pragma unroll
  for (int i = 0; i < NK; ++i) { w[i] *= inv; sm += w[i]; }
  float* o = wk + p * (NK + 1);
  constexpr int h2 = NK / 2;
#pragma unroll
  for (int i = 0; i < h2; ++i) o[i] = w[i];
  o[h2] = 1.f - sm;
#pragma unroll
  for (int i = h2; i < NK; ++i) o[i + 1] = w[i];
}

// 7) mask head: sigmoid(z+bias) * (h0 > 0.001)
__global__ void k_head_mask(const unsigned short* __restrict__ z3, int Cdst,
                            const float* __restrict__ bias,
                            const float* __restrict__ h0,
                            float* __restrict__ out3) {
  size_t p = (size_t)blockIdx.x * 256 + threadIdx.x;
  if (p >= Ptot) return;
  float valid = (h0[p] > 0.001f) ? 1.f : 0.f;
#pragma unroll
  for (int j = 0; j < 3; ++j) {
    float v = bf2f(z3[p * Cdst + j]) + bias[j];
    out3[p * 3 + j] = valid / (1.f + __expf(-v));
  }
}

// 8) softmax3 head (ck): conf = softmax(z+bias) over 3 channels
__global__ void k_head_sm3(const unsigned short* __restrict__ z3, int Cdst,
                           const float* __restrict__ bias,
                           float* __restrict__ out3) {
  size_t p = (size_t)blockIdx.x * 256 + threadIdx.x;
  if (p >= Ptot) return;
  float a0 = bf2f(z3[p * Cdst + 0]) + bias[0];
  float a1 = bf2f(z3[p * Cdst + 1]) + bias[1];
  float a2 = bf2f(z3[p * Cdst + 2]) + bias[2];
  float mx = fmaxf(a0, fmaxf(a1, a2));
  float e0 = __expf(a0 - mx), e1 = __expf(a1 - mx), e2 = __expf(a2 - mx);
  float inv = 1.f / (e0 + e1 + e2);
  out3[p * 3 + 0] = e0 * inv;
  out3[p * 3 + 1] = e1 * inv;
  out3[p * 3 + 2] = e2 * inv;
}

// ---------------------------------------------------------------------------
// 9) one CSPN propagation step for all three kernels (3/5/7), LDS halo tiles
// ---------------------------------------------------------------------------
__global__ __launch_bounds__(256) void k_cspn_step(
    const float* __restrict__ in3, const float* __restrict__ in5,
    const float* __restrict__ in7,
    const float* __restrict__ wk3, const float* __restrict__ wk5,
    const float* __restrict__ wk7,
    const float* __restrict__ mask3, const float* __restrict__ conf3,
    const float* __restrict__ h0,
    float* __restrict__ out3, float* __restrict__ out5,
    float* __restrict__ out7, float* __restrict__ mix) {
  __shared__ float t3[22 * 22];
  __shared__ float t5[22 * 22];
  __shared__ float t7[22 * 22];
  int x0 = blockIdx.x * 16, y0 = blockIdx.y * 16, b = blockIdx.z;
  int tid = threadIdx.y * 16 + threadIdx.x;
  const size_t plane = (size_t)Hh * Ww;
  const float* b3 = in3 + b * plane;
  const float* b5 = in5 + b * plane;
  const float* b7 = in7 + b * plane;
  for (int i = tid; i < 22 * 22; i += 256) {
    int ly = i / 22, lx = i % 22;
    int gy = y0 + ly - 3, gx = x0 + lx - 3;
    bool ok = (unsigned)gy < (unsigned)Hh && (unsigned)gx < (unsigned)Ww;
    size_t g = (size_t)gy * Ww + gx;
    t3[i] = ok ? b3[g] : 0.f;
    t5[i] = ok ? b5[g] : 0.f;
    t7[i] = ok ? b7[g] : 0.f;
  }
  __syncthreads();
  int x = x0 + threadIdx.x, y = y0 + threadIdx.y;
  size_t p = b * plane + (size_t)y * Ww + x;
  int cy = threadIdx.y + 3, cx = threadIdx.x + 3;

  const float* w3p = wk3 + p * 9;
  const float* w5p = wk5 + p * 25;
  const float* w7p = wk7 + p * 49;
  __builtin_prefetch((const void*)w7p, 0, 0);

  float s3 = 0.f;
#pragma unroll
  for (int k = 0; k < 9; ++k) {
    int dy = k / 3 - 1, dx = k % 3 - 1;
    s3 += w3p[k] * t3[(cy + dy) * 22 + (cx + dx)];
  }
  float s5 = 0.f;
#pragma unroll
  for (int k = 0; k < 25; ++k) {
    int dy = k / 5 - 2, dx = k % 5 - 2;
    s5 += w5p[k] * t5[(cy + dy) * 22 + (cx + dx)];
  }
  float s7 = 0.f;
#pragma unroll
  for (int k = 0; k < 49; ++k) {
    int dy = k / 7 - 3, dx = k % 7 - 3;
    s7 += w7p[k] * t7[(cy + dy) * 22 + (cx + dx)];
  }
  float h = h0[p];
  float m3 = mask3[p * 3 + 0], m5 = mask3[p * 3 + 1], m7 = mask3[p * 3 + 2];
  float o3 = (1.f - m3) * s3 + m3 * h;
  float o5 = (1.f - m5) * s5 + m5 * h;
  float o7 = (1.f - m7) * s7 + m7 * h;
  out3[p] = o3; out5[p] = o5; out7[p] = o7;
  if (mix) {
    mix[p] = conf3[p * 3 + 0] * o3 + conf3[p * 3 + 1] * o5 +
             conf3[p * 3 + 2] * o7;
  }
}

// 10) write hns = [hn, mid, fin] into ct input channels 64..66 (padded NHWC)
__global__ void k_fill_ct(const float* __restrict__ hn,
                          const float* __restrict__ mid,
                          const float* __restrict__ fin,
                          unsigned short* __restrict__ ctI) {
  size_t p = (size_t)blockIdx.x * 256 + threadIdx.x;
  if (p >= Ptot) return;
  size_t plane = (size_t)Hh * Ww;
  int b = (int)(p / plane);
  size_t rem = p - (size_t)b * plane;
  int y = (int)(rem / Ww), x = (int)(rem % Ww);
  size_t pp = ((size_t)b * Hp + y) * Wp + x;
  ctI[pp * 96 + 64] = f2bf(hn[p]);
  ctI[pp * 96 + 65] = f2bf(mid[p]);
  ctI[pp * 96 + 66] = f2bf(fin[p]);
}

// 11) final: wt = softmax(z3+bias), out = wt . [hn, mid, fin]
__global__ void k_final(const unsigned short* __restrict__ z3, int Cdst,
                        const float* __restrict__ bias,
                        const float* __restrict__ hn,
                        const float* __restrict__ mid,
                        const float* __restrict__ fin,
                        float* __restrict__ out) {
  size_t p = (size_t)blockIdx.x * 256 + threadIdx.x;
  if (p >= Ptot) return;
  float a0 = bf2f(z3[p * Cdst + 0]) + bias[0];
  float a1 = bf2f(z3[p * Cdst + 1]) + bias[1];
  float a2 = bf2f(z3[p * Cdst + 2]) + bias[2];
  float mx = fmaxf(a0, fmaxf(a1, a2));
  float e0 = __expf(a0 - mx), e1 = __expf(a1 - mx), e2 = __expf(a2 - mx);
  float inv = 1.f / (e0 + e1 + e2);
  out[p] = (e0 * hn[p] + e1 * mid[p] + e2 * fin[p]) * inv;
}

// ---------------------------------------------------------------------------
// host-side dispatch
// ---------------------------------------------------------------------------
enum GemmKind { G_C1_64, G_C1_96, G_C2_NT1, G_C2_NT2, G_C2_NT3 };

static void launch_gemm(GemmKind kind, const unsigned short* srcI,
                        const unsigned short* wf, unsigned short* dstI,
                        int Cdst, float* stats, hipStream_t s) {
  if (stats) hipMemsetAsync(stats, 0, 128 * sizeof(float), s);
  dim3 g(Bn * Hh * (Ww / 64)), b(128);
  switch (kind) {
    case G_C1_64:  k_gemm<4, 2, true ><<<g, b, 0, s>>>(srcI, wf, dstI, Cdst, stats); break;
    case G_C1_96:  k_gemm<4, 3, true ><<<g, b, 0, s>>>(srcI, wf, dstI, Cdst, stats); break;
    case G_C2_NT1: k_gemm<1, 2, false><<<g, b, 0, s>>>(srcI, wf, dstI, Cdst, stats); break;
    case G_C2_NT2: k_gemm<2, 2, false><<<g, b, 0, s>>>(srcI, wf, dstI, Cdst, stats); break;
    case G_C2_NT3: k_gemm<3, 2, false><<<g, b, 0, s>>>(srcI, wf, dstI, Cdst, stats); break;
  }
}

extern "C" void kernel_launch(void* const* d_in, const int* in_sizes, int n_in,
                              void* d_out, int out_size, void* d_ws,
                              size_t ws_size, hipStream_t stream) {
  (void)in_sizes; (void)n_in; (void)out_size; (void)ws_size;
  // --- inputs (setup_inputs dict order, params flattened in insertion order)
  const float* fout = (const float*)d_in[0];
  const float* hn   = (const float*)d_in[1];
  const float* h0   = (const float*)d_in[2];
  struct GenP { const float *c1w, *c1g, *c1b, *c2w, *c2g, *c2b; };
  GenP g3 = {(const float*)d_in[3],  (const float*)d_in[4],  (const float*)d_in[5],
             (const float*)d_in[6],  (const float*)d_in[7],  (const float*)d_in[8]};
  GenP g5 = {(const float*)d_in[9],  (const float*)d_in[10], (const float*)d_in[11],
             (const float*)d_in[12], (const float*)d_in[13], (const float*)d_in[14]};
  GenP g7 = {(const float*)d_in[15], (const float*)d_in[16], (const float*)d_in[17],
             (const float*)d_in[18], (const float*)d_in[19], (const float*)d_in[20]};
  struct HeadP { const float *c1w, *c1g, *c1b, *c2w, *c2bias; };
  HeadP hm = {(const float*)d_in[21], (const float*)d_in[22], (const float*)d_in[23],
              (const float*)d_in[24], (const float*)d_in[25]};
  HeadP hk = {(const float*)d_in[26], (const float*)d_in[27], (const float*)d_in[28],
              (const float*)d_in[29], (const float*)d_in[30]};
  HeadP ht = {(const float*)d_in[31], (const float*)d_in[32], (const float*)d_in[33],
              (const float*)d_in[34], (const float*)d_in[35]};

  // --- workspace carve-up
  char* wsb = (char*)d_ws;
  size_t cur = 0;
  auto alloc = [&](size_t bytes) -> void* {
    void* r = wsb + cur;
    cur = (cur + bytes + 255) & ~(size_t)255;
    return r;
  };
  unsigned short* NH = (unsigned short*)alloc(PadTot * 64 * 2);  // padded
  unsigned short* CT = (unsigned short*)alloc(PadTot * 96 * 2);  // padded
  unsigned short* Z1 = (unsigned short*)alloc(PadTot * 64 * 2);  // padded
  unsigned short* Z2 = (unsigned short*)alloc(Ptot * 48 * 2);    // unpadded
  unsigned short* Z3 = Z2;  // alias: never simultaneously live
  float* WK3 = (float*)alloc(Ptot * 9 * 4);
  float* WK5 = (float*)alloc(Ptot * 25 * 4);
  float* WK7 = (float*)alloc(Ptot * 49 * 4);
  float* MB  = (float*)alloc(Ptot * 3 * 4);   // mask
  float* CF  = (float*)alloc(Ptot * 3 * 4);   // conf
  float* FA3 = (float*)alloc(Ptot * 4);
  float* FA5 = (float*)alloc(Ptot * 4);
  float* FA7 = (float*)alloc(Ptot * 4);
  float* FB3 = (float*)alloc(Ptot * 4);
  float* FB5 = (float*)alloc(Ptot * 4);
  float* FB7 = (float*)alloc(Ptot * 4);
  float* MID = (float*)alloc(Ptot * 4);
  float* FIN = (float*)alloc(Ptot * 4);
  float* ST  = (float*)alloc(256 * 4);
  auto wbuf = [&](int KC, int NT) {
    return (unsigned short*)alloc((size_t)KC * NT * 32 * 16 * 2);
  };
  unsigned short* WF_g3c1 = wbuf(18, 4); unsigned short* WF_g3c2 = wbuf(18, 1);
  unsigned short* WF_g5c1 = wbuf(18, 4); unsigned short* WF_g5c2 = wbuf(18, 2);
  unsigned short* WF_g7c1 = wbuf(18, 4); unsigned short* WF_g7c2 = wbuf(18, 3);
  unsigned short* WF_mc1  = wbuf(18, 4); unsigned short* WF_mc2  = wbuf(18, 1);
  unsigned short* WF_kc1  = wbuf(18, 4); unsigned short* WF_kc2  = wbuf(18, 1);
  unsigned short* WF_tc1  = wbuf(27, 4); unsigned short* WF_tc2  = wbuf(18, 1);

  // interior-offset views (pixel (0,0) of batch 0)
  unsigned short* NHI = NH + (size_t)(Wp + 1) * 64;
  unsigned short* CTI = CT + (size_t)(Wp + 1) * 96;
  unsigned short* Z1I = Z1 + (size_t)(Wp + 1) * 64;

  const float cnt = (float)Ptot;
  dim3 eb(256);
  int gP  = (int)((Ptot + 255) / 256);
  int gPC = (int)((Ptot * 64 + 255) / 256);
  int gPB = (int)((PadTot + 255) / 256);

  // 0) border zeroing + layout conversion
  k_zero_border<<<gPB, eb, 0, stream>>>(NH, CT, Z1);
  k_convert<<<gPC, eb, 0, stream>>>(fout, NHI, CTI);

  // 2) weight swizzles
  auto swz = [&](const float* w, int Cout, int Cin, int CPT, int NT,
                 unsigned short* dst) {
    k_swizzle<<<dim3(CPT * 9 * NT), dim3(512), 0, stream>>>(w, dst, Cout, Cin,
                                                            CPT, NT);
  };
  swz(g3.c1w, 64, 64, 2, 4, WF_g3c1); swz(g3.c2w,  8, 64, 2, 1, WF_g3c2);
  swz(g5.c1w, 64, 64, 2, 4, WF_g5c1); swz(g5.c2w, 24, 64, 2, 2, WF_g5c2);
  swz(g7.c1w, 64, 64, 2, 4, WF_g7c1); swz(g7.c2w, 48, 64, 2, 3, WF_g7c2);
  swz(hm.c1w, 64, 64, 2, 4, WF_mc1);  swz(hm.c2w,  3, 64, 2, 1, WF_mc2);
  swz(hk.c1w, 64, 64, 2, 4, WF_kc1);  swz(hk.c2w,  3, 64, 2, 1, WF_kc2);
  swz(ht.c1w, 64, 67, 3, 4, WF_tc1);  swz(ht.c2w,  3, 64, 2, 1, WF_tc2);

  // 3) GenKernel branches
  auto branch = [&](const GenP& gp, unsigned short* wf1, unsigned short* wf2,
                    int NK, GemmKind k2, int Cd2, float* wk) {
    launch_gemm(G_C1_64, NHI, wf1, Z1I, 64, ST, stream);
    k_bn_finalize<<<1, 64, 0, stream>>>(ST, gp.c1g, gp.c1b, 64, cnt);
    k_bn_relu<<<gPC, eb, 0, stream>>>(Z1I, ST);
    launch_gemm(k2, Z1I, wf2, Z2, Cd2, ST, stream);
    k_bn_finalize<<<1, 64, 0, stream>>>(ST, gp.c2g, gp.c2b, NK, cnt);
    if (NK == 8)
      k_genkernel<8><<<gP, eb, 0, stream>>>(Z2, Cd2, ST, wk);
    else if (NK == 24)
      k_genkernel<24><<<gP, eb, 0, stream>>>(Z2, Cd2, ST, wk);
    else
      k_genkernel<48><<<gP, eb, 0, stream>>>(Z2, Cd2, ST, wk);
  };
  branch(g3, WF_g3c1, WF_g3c2,  8, G_C2_NT1, 16, WK3);
  branch(g5, WF_g5c1, WF_g5c2, 24, G_C2_NT2, 32, WK5);
  branch(g7, WF_g7c1, WF_g7c2, 48, G_C2_NT3, 48, WK7);

  // 4) mask + ck heads
  launch_gemm(G_C1_64, NHI, WF_mc1, Z1I, 64, ST, stream);
  k_bn_finalize<<<1, 64, 0, stream>>>(ST, hm.c1g, hm.c1b, 64, cnt);
  k_bn_relu<<<gPC, eb, 0, stream>>>(Z1I, ST);
  launch_gemm(G_C2_NT1, Z1I, WF_mc2, Z3, 16, nullptr, stream);
  k_head_mask<<<gP, eb, 0, stream>>>(Z3, 16, hm.c2bias, h0, MB);

  launch_gemm(G_C1_64, NHI, WF_kc1, Z1I, 64, ST, stream);
  k_bn_finalize<<<1, 64, 0, stream>>>(ST, hk.c1g, hk.c1b, 64, cnt);
  k_bn_relu<<<gPC, eb, 0, stream>>>(Z1I, ST);
  launch_gemm(G_C2_NT1, Z1I, WF_kc2, Z3, 16, nullptr, stream);
  k_head_sm3<<<gP, eb, 0, stream>>>(Z3, 16, hk.c2bias, CF);

  // 5) CSPN propagation (6 steps, ping-pong)
  dim3 cg(Ww / 16, Hh / 16, Bn), cb(16, 16);
  const float *i3 = hn, *i5 = hn, *i7 = hn;
  float *o3 = FA3, *o5 = FA5, *o7 = FA7;
  for (int step = 0; step < 6; ++step) {
    float* mix = (step == 2) ? MID : (step == 5) ? FIN : nullptr;
    k_cspn_step<<<cg, cb, 0, stream>>>(i3, i5, i7, WK3, WK5, WK7, MB, CF, h0,
                                       o3, o5, o7, mix);
    i3 = o3; i5 = o5; i7 = o7;
    if (o3 == FA3) { o3 = FB3; o5 = FB5; o7 = FB7; }
    else           { o3 = FA3; o5 = FA5; o7 = FA7; }
  }

  // 6) ct head on concat([fout, hn, mid, fin])
  k_fill_ct<<<gP, eb, 0, stream>>>(hn, MID, FIN, CTI);
  launch_gemm(G_C1_96, CTI, WF_tc1, Z1I, 64, ST, stream);
  k_bn_finalize<<<1, 64, 0, stream>>>(ST, ht.c1g, ht.c1b, 64, cnt);
  k_bn_relu<<<gPC, eb, 0, stream>>>(Z1I, ST);
  launch_gemm(G_C2_NT1, Z1I, WF_tc2, Z3, 16, nullptr, stream);
  k_final<<<gP, eb, 0, stream>>>(Z3, 16, ht.c2bias, hn, MID, FIN, (float*)d_out);
}
